// DeepCastle7_15771119911332
// MI455X (gfx1250) — compile-verified, matrix operand order checked by reading, not run on hardware
//
#include <hip/hip_runtime.h>

#define L1C   256
#define NBK   8
#define EMBD  264
#define BATCH 16384

typedef __attribute__((ext_vector_type(16))) _Float16 v16h;
typedef __attribute__((ext_vector_type(8)))  _Float16 v8h;
typedef __attribute__((ext_vector_type(4)))  _Float16 v4h;
typedef __attribute__((ext_vector_type(8)))  float    v8f;

__device__ __forceinline__ float clamp01(float x) { return fminf(fmaxf(x, 0.f), 1.f); }

// ---------------------------------------------------------------------------
// Pre-pack W1 (= w1 + tiled fw1) into WMMA B-matrix per-lane register layout.
// Blob layout: [bucket][kchunk(8)][ntile(2)][lane(32)][16 halfs] (32B/lane).
// B element mapping mirrors the documented 16-bit A layout with N on lanes:
//   lane<16 -> N=lane,   K in {0..7} u {16..23}
//   lane>=16-> N=lane-16,K in {8..15} u {24..31}
// ---------------------------------------------------------------------------
__global__ void prep_w1_kernel(const float* __restrict__ w1, const float* __restrict__ fw1,
                               _Float16* __restrict__ W1b) {
  int id   = blockIdx.x * blockDim.x + threadIdx.x;  // 4096 threads
  int lane = id & 31;
  int t    = (id >> 5) & 1;
  int c    = (id >> 6) & 7;
  int kb   = id >> 9;
  int kh8  = (lane >= 16) ? 8 : 0;
  int n    = t * 16 + (lane & 15);
  int row  = kb * 32 + n;  // output feature index (0..255)
  v16h v;
#pragma unroll
  for (int e = 0; e < 16; ++e) {
    int k   = (e < 8) ? (kh8 + e) : (16 + kh8 + (e - 8));
    int col = c * 32 + k;  // input feature index (0..255)
    v[e] = (_Float16)(w1[row * 256 + col] + fw1[n * 256 + col]);
  }
  *(v16h*)(W1b + (size_t)id * 16) = v;
}

// W2 blob: [bucket][kchunk(2)][ntile(2)][lane(32)][16], K padded 62 -> 64.
__global__ void prep_w2_kernel(const float* __restrict__ w2, _Float16* __restrict__ W2b) {
  int id   = blockIdx.x * blockDim.x + threadIdx.x;  // 1024 threads
  int lane = id & 31;
  int t    = (id >> 5) & 1;
  int c    = (id >> 6) & 1;
  int kb   = id >> 7;
  int kh8  = (lane >= 16) ? 8 : 0;
  int n    = t * 16 + (lane & 15);
  int row  = kb * 32 + n;
  v16h v;
#pragma unroll
  for (int e = 0; e < 16; ++e) {
    int k   = (e < 8) ? (kh8 + e) : (16 + kh8 + (e - 8));
    int col = c * 32 + k;
    v[e] = (col < 62) ? (_Float16)w2[row * 62 + col] : (_Float16)0.0f;
  }
  *(v16h*)(W2b + (size_t)id * 16) = v;
}

// ---------------------------------------------------------------------------
// Feature transform: gather-sum 2x32 embedding rows, clipped mix, pairwise
// product, emit f16 activations (B x 256) + per-sample psqt contribution.
// 64 threads per sample (coalesced float4 row reads), 4 samples per block.
// ---------------------------------------------------------------------------
__global__ void __launch_bounds__(256) ft_kernel(
    const float* __restrict__ emb,
    const int* __restrict__ w_idx, const int* __restrict__ b_idx,
    const float* __restrict__ us, const float* __restrict__ them,
    const int* __restrict__ pcnt,
    _Float16* __restrict__ X, float* __restrict__ psqt) {
  __shared__ float lds0[4][512];
  __shared__ float ldsP[4][32];
  int sl = threadIdx.x >> 6;
  int t  = threadIdx.x & 63;
  int s  = blockIdx.x * 4 + sl;

  const int* wi = w_idx + (size_t)s * 32;
  const int* bi = b_idx + (size_t)s * 32;

  float4 aw = make_float4(0.f, 0.f, 0.f, 0.f);
  float4 ab = make_float4(0.f, 0.f, 0.f, 0.f);
#pragma unroll 4
  for (int j = 0; j < 32; ++j) {
    float4 a = *((const float4*)(emb + (size_t)wi[j] * EMBD) + t);
    float4 b = *((const float4*)(emb + (size_t)bi[j] * EMBD) + t);
    aw.x += a.x; aw.y += a.y; aw.z += a.z; aw.w += a.w;
    ab.x += b.x; ab.y += b.y; ab.z += b.z; ab.w += b.w;
  }

  int bk; { int p = pcnt[s]; bk = (p - 1) >> 2; bk = bk < 0 ? 0 : (bk > 7 ? 7 : bk); }
  float u = us[s], th = them[s];

  if (t < 32) {  // psqt column (only bucket column is needed)
    float pw = emb[(size_t)wi[t] * EMBD + 256 + bk];
    float pb = emb[(size_t)bi[t] * EMBD + 256 + bk];
    ldsP[sl][t] = pw - pb;
  }

  float w4[4] = {aw.x, aw.y, aw.z, aw.w};
  float b4[4] = {ab.x, ab.y, ab.z, ab.w};
#pragma unroll
  for (int k = 0; k < 4; ++k) {
    lds0[sl][4 * t + k]       = clamp01(u * w4[k] + th * b4[k]);  // l0[0:256]
    lds0[sl][256 + 4 * t + k] = clamp01(u * b4[k] + th * w4[k]);  // l0[256:512]
  }
  __syncthreads();

  const float SC = 127.f / 128.f;
  v4h o;
  if (t < 32) {
#pragma unroll
    for (int k = 0; k < 4; ++k) {
      int j = 4 * t + k;
      o[k] = (_Float16)(lds0[sl][j] * lds0[sl][j + 128] * SC);
    }
    *(v4h*)(X + (size_t)s * 256 + 4 * t) = o;
  } else {
    int tt = t - 32;
#pragma unroll
    for (int k = 0; k < 4; ++k) {
      int j = 4 * tt + k;
      o[k] = (_Float16)(lds0[sl][256 + j] * lds0[sl][384 + j] * SC);
    }
    *(v4h*)(X + (size_t)s * 256 + 128 + 4 * tt) = o;
  }
  if (t == 0) {
    float sum = 0.f;
#pragma unroll
    for (int j = 0; j < 32; ++j) sum += ldsP[sl][j];
    psqt[s] = sum * (u - 0.5f);
  }
}

// ---------------------------------------------------------------------------
// MLP: one wave32 per tile of 16 samples. Dense-over-buckets WMMA + per-row
// bucket masking (avoids a data-dependent compaction pass; extra FLOPs are
// negligible vs matrix-core throughput).
// ---------------------------------------------------------------------------
__global__ void __launch_bounds__(32) mlp_kernel(
    const _Float16* __restrict__ X, const float* __restrict__ psqt,
    const _Float16* __restrict__ W1b, const _Float16* __restrict__ W2b,
    const float* __restrict__ b1, const float* __restrict__ fb1,
    const float* __restrict__ b2, const float* __restrict__ wo,
    const float* __restrict__ bo, const int* __restrict__ pcnt,
    float* __restrict__ out) {
  __shared__ float ldsB1[256];
  __shared__ int   ldsBkt[16];
  __shared__ float ldsL1[16 * 32];
  __shared__ float ldsL2[16 * 32];

  int lane = threadIdx.x;
  int base = blockIdx.x * 16;

  for (int i = lane; i < 256; i += 32) ldsB1[i] = b1[i] + fb1[i & 31];
  if (lane < 16) {
    int p = pcnt[base + lane];
    int bk = (p - 1) >> 2; bk = bk < 0 ? 0 : (bk > 7 ? 7 : bk);
    ldsBkt[lane] = bk;
  }
  __syncthreads();

  unsigned bm = 0;
  for (int m = 0; m < 16; ++m) bm |= 1u << ldsBkt[m];
  bm = (unsigned)__builtin_amdgcn_readfirstlane((int)bm);  // force SGPR branch

  int kh8  = (lane >= 16) ? 8 : 0;
  int moff = (lane >= 16) ? 8 : 0;  // D-layout row offset for this lane half
  int n0   = lane & 15;             // D-layout column / A-layout row

  // A tile (16x256 f16), WMMA A-register layout: two contiguous v8h per chunk.
  v16h A[8];
  const _Float16* xr = X + (size_t)(base + n0) * 256;
#pragma unroll
  for (int c = 0; c < 8; ++c) {
    v8h lo = *(const v8h*)(xr + c * 32 + kh8);
    v8h hi = *(const v8h*)(xr + c * 32 + 16 + kh8);
    v16h a;
#pragma unroll
    for (int e = 0; e < 8; ++e) { a[e] = lo[e]; a[8 + e] = hi[e]; }
    A[c] = a;
  }

  // ---- layer 1: 256 -> 32, all present buckets, per-row select ----
  v8f sel0 = {}; v8f sel1 = {};
  for (int kb = 0; kb < 8; ++kb) {
    if (!((bm >> kb) & 1u)) continue;
    v8f acc0 = {}; v8f acc1 = {};
    const v16h* Bp = (const v16h*)W1b + (size_t)kb * 8 * 2 * 32;
#pragma unroll
    for (int c = 0; c < 8; ++c) {
      v16h bv0 = Bp[(c * 2 + 0) * 32 + lane];
      v16h bv1 = Bp[(c * 2 + 1) * 32 + lane];
      acc0 = __builtin_amdgcn_wmma_f32_16x16x32_f16(false, A[c], false, bv0,
                                                    (short)0, acc0, false, false);
      acc1 = __builtin_amdgcn_wmma_f32_16x16x32_f16(false, A[c], false, bv1,
                                                    (short)0, acc1, false, false);
    }
#pragma unroll
    for (int r = 0; r < 8; ++r) {
      bool hit = (ldsBkt[r + moff] == kb);
      sel0[r] = hit ? acc0[r] : sel0[r];
      sel1[r] = hit ? acc1[r] : sel1[r];
    }
  }
#pragma unroll
  for (int r = 0; r < 8; ++r) {
    int m = r + moff;
    int bk = ldsBkt[m];
    ldsL1[m * 32 + n0]      = sel0[r] + ldsB1[bk * 32 + n0];
    ldsL1[m * 32 + 16 + n0] = sel1[r] + ldsB1[bk * 32 + 16 + n0];
  }
  __syncthreads();

  // ---- layer 2 input: [clip(x^2*255/256) (31) | clip(x) (31) | 0 0], K=64 ----
  v16h A2[2];
#pragma unroll
  for (int c = 0; c < 2; ++c) {
    v16h a;
#pragma unroll
    for (int e = 0; e < 16; ++e) {
      int k = c * 32 + ((e < 8) ? (kh8 + e) : (16 + kh8 + (e - 8)));
      float v;
      if (k < 31)      { float x = ldsL1[n0 * 32 + k];        v = fminf(x * x * (255.f / 256.f), 1.f); }
      else if (k < 62) { float x = ldsL1[n0 * 32 + (k - 31)]; v = clamp01(x); }
      else             v = 0.f;
      a[e] = (_Float16)v;
    }
    A2[c] = a;
  }

  v8f s20 = {}; v8f s21 = {};
  for (int kb = 0; kb < 8; ++kb) {
    if (!((bm >> kb) & 1u)) continue;
    v8f acc0 = {}; v8f acc1 = {};
    const v16h* Bp = (const v16h*)W2b + (size_t)kb * 2 * 2 * 32;
#pragma unroll
    for (int c = 0; c < 2; ++c) {
      v16h bv0 = Bp[(c * 2 + 0) * 32 + lane];
      v16h bv1 = Bp[(c * 2 + 1) * 32 + lane];
      acc0 = __builtin_amdgcn_wmma_f32_16x16x32_f16(false, A2[c], false, bv0,
                                                    (short)0, acc0, false, false);
      acc1 = __builtin_amdgcn_wmma_f32_16x16x32_f16(false, A2[c], false, bv1,
                                                    (short)0, acc1, false, false);
    }
#pragma unroll
    for (int r = 0; r < 8; ++r) {
      bool hit = (ldsBkt[r + moff] == kb);
      s20[r] = hit ? acc0[r] : s20[r];
      s21[r] = hit ? acc1[r] : s21[r];
    }
  }
#pragma unroll
  for (int r = 0; r < 8; ++r) {
    int m = r + moff;
    int bk = ldsBkt[m];
    ldsL2[m * 32 + n0]      = clamp01(s20[r] + b2[bk * 32 + n0]);
    ldsL2[m * 32 + 16 + n0] = clamp01(s21[r] + b2[bk * 32 + 16 + n0]);
  }
  __syncthreads();

  // ---- layer 3 (32 -> 1) + l1x_out skip + psqt ----
  if (lane < 16) {
    int m = lane;
    int bk = ldsBkt[m];
    float s = bo[bk];
#pragma unroll 8
    for (int n = 0; n < 32; ++n) s += ldsL2[m * 32 + n] * wo[bk * 32 + n];
    s += ldsL1[m * 32 + 31];   // l1x_out (raw l1c column 31)
    s += psqt[base + m];
    out[base + m] = s;
  }
}

extern "C" void kernel_launch(void* const* d_in, const int* in_sizes, int n_in,
                              void* d_out, int out_size, void* d_ws, size_t ws_size,
                              hipStream_t stream) {
  const float* emb  = (const float*)d_in[0];
  const float* w1   = (const float*)d_in[1];
  const float* b1   = (const float*)d_in[2];
  const float* fw1  = (const float*)d_in[3];
  const float* fb1  = (const float*)d_in[4];
  const float* w2   = (const float*)d_in[5];
  const float* b2   = (const float*)d_in[6];
  const float* wo   = (const float*)d_in[7];
  const float* bo   = (const float*)d_in[8];
  const float* us   = (const float*)d_in[9];
  const float* them = (const float*)d_in[10];
  const int* w_idx  = (const int*)d_in[11];
  const int* b_idx  = (const int*)d_in[12];
  const int* pcnt   = (const int*)d_in[13];
  float* out = (float*)d_out;

  char* ws = (char*)d_ws;
  size_t offX  = 0;
  size_t offP  = offX + (size_t)BATCH * 256 * sizeof(_Float16);  // 8 MB activations
  size_t offW1 = offP + (size_t)BATCH * sizeof(float);           // psqt
  size_t offW2 = offW1 + (size_t)NBK * 8 * 2 * 32 * 16 * sizeof(_Float16);
  _Float16* X    = (_Float16*)(ws + offX);
  float*    psqt = (float*)(ws + offP);
  _Float16* W1b  = (_Float16*)(ws + offW1);
  _Float16* W2b  = (_Float16*)(ws + offW2);

  prep_w1_kernel<<<16, 256, 0, stream>>>(w1, fw1, W1b);
  prep_w2_kernel<<<4, 256, 0, stream>>>(w2, W2b);
  ft_kernel<<<BATCH / 4, 256, 0, stream>>>(emb, w_idx, b_idx, us, them, pcnt, X, psqt);
  mlp_kernel<<<BATCH / 16, 32, 0, stream>>>(X, psqt, W1b, W2b, b1, fb1, b2, wo, bo, pcnt, out);
}
